// Block_3453153706730
// MI455X (gfx1250) — matrix operand order repaired and checked by reading, not compiled, not run on hardware
//
#include <hip/hip_runtime.h>
#include <hip/hip_bf16.h>

// ---------------------------------------------------------------------------
// Types
// ---------------------------------------------------------------------------
typedef __bf16 bf16;
typedef __bf16 v16bf __attribute__((ext_vector_type(16)));
typedef float  v8f   __attribute__((ext_vector_type(8)));

__device__ __forceinline__ bf16 f2bf(float f) {
  union { float f; unsigned u; } a; a.f = f;
  unsigned r = a.u + 0x7FFFu + ((a.u >> 16) & 1u);   // round-to-nearest-even
  unsigned short h = (unsigned short)(r >> 16);
  bf16 b; __builtin_memcpy(&b, &h, 2); return b;
}

__device__ __forceinline__ v8f wmma_bf16(v16bf a, v16bf b, v8f c) {
  return __builtin_amdgcn_wmma_f32_16x16x32_bf16(false, a, false, b, (short)0, c,
                                                 false, false);
}

// A fragment: 16x32 bf16, row-major A[M x K] (global or LDS).
// lanes 0-15 -> M=lane, K=0..15 contiguous;
// lanes 16-31 same rows, K=16..31 (matches load_b_frag).
__device__ __forceinline__ v16bf load_a_frag(const bf16* A, int lda, int row,
                                             int kk, int lane) {
  const int khalf = (lane & 16) ? 16 : 0;
  const bf16* p = A + (size_t)row * lda + kk + khalf;
  v16bf a;
#pragma unroll
  for (int j = 0; j < 8; ++j) {
    const int kj = 2 * j;
    unsigned d; __builtin_memcpy(&d, p + kj, 4);      // two packed bf16
    unsigned short l16 = (unsigned short)(d & 0xFFFFu);
    unsigned short h16 = (unsigned short)(d >> 16);
    bf16 lo, hi; __builtin_memcpy(&lo, &l16, 2); __builtin_memcpy(&hi, &h16, 2);
    a[2 * j] = lo; a[2 * j + 1] = hi;
  }
  return a;
}

// B fragment: 32x16 bf16 from B^T[N x K] (contiguous 16 K-values per lane).
// lanes 0-15: K=kk..kk+15 of column (lane), lanes 16-31: K=kk+16..kk+31.
__device__ __forceinline__ v16bf load_b_frag(const bf16* Bt, int ldb, int col,
                                             int kk, int lane) {
  return *reinterpret_cast<const v16bf*>(
      Bt + (size_t)col * ldb + kk + ((lane & 16) ? 16 : 0));
}

// ---------------------------------------------------------------------------
// Async global->LDS staging (CDNA5 ASYNCcnt path).
// Stages a 64-row x 32-col bf16 tile (4 KB): 256 16-byte chunks, 128 threads,
// 2 GLOBAL_LOAD_ASYNC_TO_LDS_B128 per thread. Tile row stride in LDS = 64 B.
// ---------------------------------------------------------------------------
__device__ __forceinline__ void stage_tile_async(const bf16* __restrict__ gsrc,
                                                 int ld_elems, int row0, int kk,
                                                 bf16* lds_tile, int tid) {
#pragma unroll
  for (int r = 0; r < 2; ++r) {
    const int c = tid + r * 128;             // chunk id 0..255
    const int row = c >> 2;                  // 4 chunks per 64 B row
    const int kb = (c & 3) * 16;             // byte offset within row
    const char* g =
        (const char*)(gsrc + (size_t)(row0 + row) * ld_elems + kk) + kb;
    const unsigned lds_off = (unsigned)(size_t)((char*)lds_tile + c * 16);
    asm volatile("global_load_async_to_lds_b128 %0, %1, off"
                 :: "v"(lds_off), "v"(g) : "memory");
  }
}

#define WAIT_ASYNC_LE(n) asm volatile("s_wait_asynccnt " #n ::: "memory")

// ---------------------------------------------------------------------------
// K0: float32 W[K x N] -> bf16 W^T[N x K]
// ---------------------------------------------------------------------------
__global__ void k_transpose_bf16(const float* __restrict__ W, bf16* __restrict__ Wt,
                                 int K, int N) {
  size_t total = (size_t)K * N;
  for (size_t i = (size_t)blockIdx.x * blockDim.x + threadIdx.x; i < total;
       i += (size_t)gridDim.x * blockDim.x) {
    size_t k = i / N, n = i % N;
    Wt[n * (size_t)K + k] = f2bf(W[i]);
  }
}

// ---------------------------------------------------------------------------
// K1/K6: LayerNorm over C=768, one row per block, bf16 output
// ---------------------------------------------------------------------------
__global__ __launch_bounds__(256) void k_layernorm(const float* __restrict__ X,
                                                   const float* __restrict__ g,
                                                   const float* __restrict__ b,
                                                   bf16* __restrict__ Y, int C) {
  const size_t row = blockIdx.x;
  const float* x = X + row * C;
  bf16* y = Y + row * C;
  float s = 0.f, s2 = 0.f;
  for (int c = threadIdx.x; c < C; c += blockDim.x) {
    float v = x[c]; s += v; s2 += v * v;
  }
  __shared__ float rs[256], rq[256];
  rs[threadIdx.x] = s; rq[threadIdx.x] = s2; __syncthreads();
  for (int st = 128; st > 0; st >>= 1) {
    if ((int)threadIdx.x < st) { rs[threadIdx.x] += rs[threadIdx.x + st];
                                 rq[threadIdx.x] += rq[threadIdx.x + st]; }
    __syncthreads();
  }
  float mu = rs[0] / C;
  float var = rq[0] / C - mu * mu;
  float rstd = rsqrtf(var + 1e-5f);
  for (int c = threadIdx.x; c < C; c += blockDim.x)
    y[c] = f2bf((x[c] - mu) * rstd * g[c] + b[c]);
}

// ---------------------------------------------------------------------------
// K2/K5/K8: bf16 WMMA GEMM, f32 out, optional residual.
// Block = 4 waves = 64x64 output tile. Double-buffered async LDS staging:
// each K-step stages the next 64x32 A and B tiles with
// GLOBAL_LOAD_ASYNC_TO_LDS_B128 while computing from the current buffers.
// grid = (N/64, M/64)
// ---------------------------------------------------------------------------
__global__ __launch_bounds__(128) void k_gemm_f32(const bf16* __restrict__ A,
                                                  const bf16* __restrict__ Bt,
                                                  const float* __restrict__ residual,
                                                  float* __restrict__ Cout,
                                                  int M, int N, int K) {
  __shared__ __attribute__((aligned(64))) bf16 sA[2][64 * 32];
  __shared__ __attribute__((aligned(64))) bf16 sB[2][64 * 32];
  const int tid = threadIdx.x;
  const int lane = tid & 31;
  const int wave = tid >> 5;
  const int row0 = blockIdx.y * 64;
  const int col0 = blockIdx.x * 64;

  v8f acc0 = {0.f,0.f,0.f,0.f,0.f,0.f,0.f,0.f};
  v8f acc1 = acc0, acc2 = acc0, acc3 = acc0;

  stage_tile_async(A,  K, row0, 0, sA[0], tid);
  stage_tile_async(Bt, K, col0, 0, sB[0], tid);
  int cur = 0;
  for (int kk = 0; kk < K; kk += 32) {
    const bool more = (kk + 32) < K;
    if (more) {                                   // prefetch next K-step
      stage_tile_async(A,  K, row0, kk + 32, sA[cur ^ 1], tid);
      stage_tile_async(Bt, K, col0, kk + 32, sB[cur ^ 1], tid);
      WAIT_ASYNC_LE(0x4);                         // current 4 async ops done
    } else {
      WAIT_ASYNC_LE(0x0);
    }
    __syncthreads();                              // publish LDS across waves
    v16bf a = load_a_frag(sA[cur], 32, wave * 16 + (lane & 15), 0, lane);
    acc0 = wmma_bf16(a, load_b_frag(sB[cur], 32,  0 + (lane & 15), 0, lane), acc0);
    acc1 = wmma_bf16(a, load_b_frag(sB[cur], 32, 16 + (lane & 15), 0, lane), acc1);
    acc2 = wmma_bf16(a, load_b_frag(sB[cur], 32, 32 + (lane & 15), 0, lane), acc2);
    acc3 = wmma_bf16(a, load_b_frag(sB[cur], 32, 48 + (lane & 15), 0, lane), acc3);
    __syncthreads();                              // done reading before refill
    cur ^= 1;
  }
  const int nn = lane & 15;
  const int mh = (lane >> 4) * 8;
  v8f accs[4] = {acc0, acc1, acc2, acc3};
#pragma unroll
  for (int s = 0; s < 4; ++s) {
#pragma unroll
    for (int i = 0; i < 8; ++i) {
      size_t idx = (size_t)(row0 + wave * 16 + mh + i) * N + (col0 + s * 16 + nn);
      float v = accs[s][i];
      if (residual) v += residual[idx];
      Cout[idx] = v;
    }
  }
}

// ---------------------------------------------------------------------------
// K3: qkv postprocess: per-head LN of q and k (hd=64), scale q, pack q/k/v bf16
// qkv f32 layout [b, n, s(3), h(12), d(64)]; outputs [b, h, n, d]
// grid = B*H*N blocks, 64 threads
// ---------------------------------------------------------------------------
__global__ __launch_bounds__(64) void k_qkv_post(const float* __restrict__ QKV,
                                                 const float* __restrict__ qg,
                                                 const float* __restrict__ qb,
                                                 const float* __restrict__ kg,
                                                 const float* __restrict__ kb,
                                                 bf16* __restrict__ Q,
                                                 bf16* __restrict__ Kv,
                                                 bf16* __restrict__ V) {
  const int id = blockIdx.x;
  const int n = id % 65;
  const int bh = id / 65;
  const int h = bh % 12;
  const int b = bh / 12;
  const int d = threadIdx.x;
  const size_t rowbase = ((size_t)(b * 65 + n)) * 2304 + h * 64 + d;
  float qv = QKV[rowbase];
  float kv = QKV[rowbase + 768];
  float vv = QKV[rowbase + 1536];
  __shared__ float sm[64];
  auto bsum = [&](float v) -> float {
    sm[d] = v; __syncthreads();
    for (int st = 32; st > 0; st >>= 1) { if (d < st) sm[d] += sm[d + st]; __syncthreads(); }
    float r = sm[0]; __syncthreads(); return r;
  };
  float qs = bsum(qv), qs2 = bsum(qv * qv);
  float qmu = qs * (1.f / 64.f), qr = rsqrtf(qs2 * (1.f / 64.f) - qmu * qmu + 1e-5f);
  float ks = bsum(kv), ks2 = bsum(kv * kv);
  float kmu = ks * (1.f / 64.f), kr = rsqrtf(ks2 * (1.f / 64.f) - kmu * kmu + 1e-5f);
  const size_t o = (((size_t)bh) * 65 + n) * 64 + d;
  Q[o]  = f2bf(((qv - qmu) * qr * qg[d] + qb[d]) * 0.125f);  // 64^-0.5
  Kv[o] = f2bf((kv - kmu) * kr * kg[d] + kb[d]);
  V[o]  = f2bf(vv);
}

// ---------------------------------------------------------------------------
// K4: attention, one (b,h) per block, 4 waves, fully LDS-resident.
// S = Q K^T (WMMA, padded 65->80), softmax(+bias), O = P V (WMMA, K padded to 96)
// LDS layout: [0,20480)=Q,K then aliased by P; [20480,46080)=S f32; [46080,58368)=V^T
// ---------------------------------------------------------------------------
__global__ __launch_bounds__(128) void k_attention(const bf16* __restrict__ Q,
                                                   const bf16* __restrict__ Kv,
                                                   const bf16* __restrict__ V,
                                                   const float* __restrict__ bias,
                                                   const float* __restrict__ bscale_p,
                                                   bf16* __restrict__ O) {
  __shared__ __attribute__((aligned(32))) char smem[58368];
  bf16*  sQ  = (bf16*)(smem);             // [80][64]
  bf16*  sK  = (bf16*)(smem + 10240);     // [80][64]
  bf16*  sP  = (bf16*)(smem);             // [80][96], aliases Q/K after sync
  float* sS  = (float*)(smem + 20480);    // [80][80]
  bf16*  sVt = (bf16*)(smem + 46080);     // [64][96]  (V transposed, K-dim padded)

  const int bh = blockIdx.x;
  const int h = bh % 12;
  const int b = bh / 12;
  const int tid = threadIdx.x;
  const int lane = tid & 31;
  const int wave = tid >> 5;
  const bf16* qp = Q  + (size_t)bh * 65 * 64;
  const bf16* kp = Kv + (size_t)bh * 65 * 64;
  const bf16* vp = V  + (size_t)bh * 65 * 64;
  const bf16 z = f2bf(0.f);

  for (int i = tid; i < 80 * 64; i += 128) {
    int n = i >> 6, d = i & 63;
    sQ[i] = (n < 65) ? qp[n * 64 + d] : z;
    sK[i] = (n < 65) ? kp[n * 64 + d] : z;
  }
  for (int i = tid; i < 96 * 64; i += 128) {
    int c = i >> 6, d = i & 63;
    sVt[d * 96 + c] = (c < 65) ? vp[c * 64 + d] : z;   // transpose into LDS
  }
  __syncthreads();

  // ---- S = Q K^T : 5x5 tiles of 16x16, K-dim = 64 (2 WMMA steps) ----
  for (int t = wave; t < 25; t += 4) {
    const int m0 = (t / 5) * 16, n0 = (t % 5) * 16;
    v8f acc = {0.f,0.f,0.f,0.f,0.f,0.f,0.f,0.f};
#pragma unroll
    for (int kk = 0; kk < 64; kk += 32) {
      v16bf a = load_a_frag(sQ, 64, m0 + (lane & 15), kk, lane);
      v16bf bb = load_b_frag(sK, 64, n0 + (lane & 15), kk, lane);  // K^T columns = K rows
      acc = wmma_bf16(a, bb, acc);
    }
    const int nn = lane & 15, mh = (lane >> 4) * 8;
#pragma unroll
    for (int i = 0; i < 8; ++i) sS[(m0 + mh + i) * 80 + n0 + nn] = acc[i];
  }
  __syncthreads();

  // ---- softmax rows (65 valid cols), + bias*scale, write P bf16 (padded) ----
  const float bscale = *bscale_p;
  for (int r = tid; r < 65; r += 128) {
    const float* brow = bias + ((size_t)h * 65 + r) * 65;
    float mx = -3.4e38f;
    for (int c = 0; c < 65; ++c) {
      float v = sS[r * 80 + c] + brow[c] * bscale;
      sS[r * 80 + c] = v; mx = fmaxf(mx, v);
    }
    float sum = 0.f;
    for (int c = 0; c < 65; ++c) {
      float e = __expf(sS[r * 80 + c] - mx);
      sS[r * 80 + c] = e; sum += e;
    }
    float inv = 1.f / sum;
    for (int c = 0; c < 96; ++c)
      sP[r * 96 + c] = (c < 65) ? f2bf(sS[r * 80 + c] * inv) : z;
  }
  __syncthreads();

  // ---- O = P V : 5x4 tiles, K-dim padded to 96 (3 WMMA steps) ----
  for (int t = wave; t < 20; t += 4) {
    const int m0 = (t / 4) * 16, d0 = (t % 4) * 16;
    v8f acc = {0.f,0.f,0.f,0.f,0.f,0.f,0.f,0.f};
#pragma unroll
    for (int kk = 0; kk < 96; kk += 32) {
      v16bf a = load_a_frag(sP, 96, m0 + (lane & 15), kk, lane);
      v16bf bb = load_b_frag(sVt, 96, d0 + (lane & 15), kk, lane);
      acc = wmma_bf16(a, bb, acc);
    }
    const int nn = lane & 15, mh = (lane >> 4) * 8;
#pragma unroll
    for (int i = 0; i < 8; ++i) {
      int m = m0 + mh + i;
      if (m < 65)   // o[b][n][h][d] = [B,N,C] for the proj GEMM
        O[(((size_t)b * 65 + m) * 12 + h) * 64 + d0 + nn] = f2bf(acc[i]);
    }
  }
}

// ---------------------------------------------------------------------------
// K7: fused gate/value GEMM + SiLU: G = (H2 Wg) * silu(H2 Wv), bf16 out.
// Same async double-buffered scheme; stages A + both weight tiles (6 async/step).
// grid = (3072/64, M/64), block = 128
// ---------------------------------------------------------------------------
__global__ __launch_bounds__(128) void k_swiglu(const bf16* __restrict__ A,
                                                const bf16* __restrict__ WgT,
                                                const bf16* __restrict__ WvT,
                                                bf16* __restrict__ G,
                                                int M, int N, int K) {
  __shared__ __attribute__((aligned(64))) bf16 sA[2][64 * 32];
  __shared__ __attribute__((aligned(64))) bf16 sG[2][64 * 32];
  __shared__ __attribute__((aligned(64))) bf16 sV[2][64 * 32];
  const int tid = threadIdx.x;
  const int lane = tid & 31;
  const int wave = tid >> 5;
  const int row0 = blockIdx.y * 64;
  const int col0 = blockIdx.x * 64;

  v8f zero = {0.f,0.f,0.f,0.f,0.f,0.f,0.f,0.f};
  v8f ag[4] = {zero, zero, zero, zero};
  v8f av[4] = {zero, zero, zero, zero};

  stage_tile_async(A,   K, row0, 0, sA[0], tid);
  stage_tile_async(WgT, K, col0, 0, sG[0], tid);
  stage_tile_async(WvT, K, col0, 0, sV[0], tid);
  int cur = 0;
  for (int kk = 0; kk < K; kk += 32) {
    const bool more = (kk + 32) < K;
    if (more) {
      stage_tile_async(A,   K, row0, kk + 32, sA[cur ^ 1], tid);
      stage_tile_async(WgT, K, col0, kk + 32, sG[cur ^ 1], tid);
      stage_tile_async(WvT, K, col0, kk + 32, sV[cur ^ 1], tid);
      WAIT_ASYNC_LE(0x6);
    } else {
      WAIT_ASYNC_LE(0x0);
    }
    __syncthreads();
    v16bf a = load_a_frag(sA[cur], 32, wave * 16 + (lane & 15), 0, lane);
#pragma unroll
    for (int s = 0; s < 4; ++s) {
      ag[s] = wmma_bf16(a, load_b_frag(sG[cur], 32, s * 16 + (lane & 15), 0, lane), ag[s]);
      av[s] = wmma_bf16(a, load_b_frag(sV[cur], 32, s * 16 + (lane & 15), 0, lane), av[s]);
    }
    __syncthreads();
    cur ^= 1;
  }
  const int nn = lane & 15, mh = (lane >> 4) * 8;
#pragma unroll
  for (int s = 0; s < 4; ++s) {
#pragma unroll
    for (int i = 0; i < 8; ++i) {
      float gv = ag[s][i], vv = av[s][i];
      float silu = vv / (1.f + __expf(-vv));
      G[(size_t)(row0 + wave * 16 + mh + i) * N + (col0 + s * 16 + nn)] = f2bf(gv * silu);
    }
  }
}

// ---------------------------------------------------------------------------
// Host launcher
// ---------------------------------------------------------------------------
extern "C" void kernel_launch(void* const* d_in, const int* in_sizes, int n_in,
                              void* d_out, int out_size, void* d_ws, size_t ws_size,
                              hipStream_t stream) {
  (void)in_sizes; (void)n_in; (void)out_size; (void)ws_size;
  const float* x        = (const float*)d_in[0];
  const float* n1_g     = (const float*)d_in[1];
  const float* n1_b     = (const float*)d_in[2];
  const float* w_qkv    = (const float*)d_in[3];
  const float* qn_g     = (const float*)d_in[4];
  const float* qn_b     = (const float*)d_in[5];
  const float* kn_g     = (const float*)d_in[6];
  const float* kn_b     = (const float*)d_in[7];
  const float* attnbias = (const float*)d_in[8];
  const float* bscale   = (const float*)d_in[9];
  const float* w_proj   = (const float*)d_in[10];
  const float* n2_g     = (const float*)d_in[11];
  const float* n2_b     = (const float*)d_in[12];
  const float* w_gate   = (const float*)d_in[13];
  const float* w_value  = (const float*)d_in[14];
  const float* w_out    = (const float*)d_in[15];
  float* out = (float*)d_out;

  const size_t M = 1024ull * 65;   // 66560 tokens (divisible by 64)
  char* ws = (char*)d_ws;
  // workspace layout (bytes); qkv f32 region is reused for G, hbf reused for h2
  size_t o_hbf   = 0;                           // M*768  bf16 (h, then h2)
  size_t o_qkv   = o_hbf   + M * 768 * 2;       // M*2304 f32  (later: G, M*3072 bf16)
  size_t o_q     = o_qkv   + M * 2304 * 4;      // M*768  bf16
  size_t o_k     = o_q     + M * 768 * 2;
  size_t o_v     = o_k     + M * 768 * 2;
  size_t o_obf   = o_v     + M * 768 * 2;       // attention output bf16
  size_t o_xres  = o_obf   + M * 768 * 2;       // M*768 f32 (x + attn)
  size_t o_wqkv  = o_xres  + M * 768 * 4;       // bf16 transposed weights
  size_t o_wproj = o_wqkv  + 768ull * 2304 * 2;
  size_t o_wgate = o_wproj + 768ull * 768 * 2;
  size_t o_wval  = o_wgate + 768ull * 3072 * 2;
  size_t o_wout  = o_wval  + 768ull * 3072 * 2;

  bf16* hbf   = (bf16*)(ws + o_hbf);
  float* qkvf = (float*)(ws + o_qkv);
  bf16* qb    = (bf16*)(ws + o_q);
  bf16* kb    = (bf16*)(ws + o_k);
  bf16* vb    = (bf16*)(ws + o_v);
  bf16* obf   = (bf16*)(ws + o_obf);
  float* xres = (float*)(ws + o_xres);
  bf16* gbuf  = (bf16*)(ws + o_qkv);            // reuse qkv region
  bf16* wqkvT  = (bf16*)(ws + o_wqkv);
  bf16* wprojT = (bf16*)(ws + o_wproj);
  bf16* wgateT = (bf16*)(ws + o_wgate);
  bf16* wvalT  = (bf16*)(ws + o_wval);
  bf16* woutT  = (bf16*)(ws + o_wout);

  // K0: weights -> bf16, transposed [N x K]
  k_transpose_bf16<<<2048, 256, 0, stream>>>(w_qkv,   wqkvT,  768, 2304);
  k_transpose_bf16<<<2048, 256, 0, stream>>>(w_proj,  wprojT, 768, 768);
  k_transpose_bf16<<<2048, 256, 0, stream>>>(w_gate,  wgateT, 768, 3072);
  k_transpose_bf16<<<2048, 256, 0, stream>>>(w_value, wvalT,  768, 3072);
  k_transpose_bf16<<<2048, 256, 0, stream>>>(w_out,   woutT,  3072, 768);

  // K1: h = LN1(x) -> bf16
  k_layernorm<<<(unsigned)M, 256, 0, stream>>>(x, n1_g, n1_b, hbf, 768);

  // K2: qkv = h @ w_qkv  (f32 out)
  k_gemm_f32<<<dim3(2304 / 64, (unsigned)(M / 64)), 128, 0, stream>>>(
      hbf, wqkvT, nullptr, qkvf, (int)M, 2304, 768);

  // K3: per-head LN of q,k (+scale), pack q/k/v bf16 as [B,H,N,hd]
  k_qkv_post<<<1024 * 12 * 65, 64, 0, stream>>>(qkvf, qn_g, qn_b, kn_g, kn_b,
                                                qb, kb, vb);

  // K4: attention per (b,h)
  k_attention<<<1024 * 12, 128, 0, stream>>>(qb, kb, vb, attnbias, bscale, obf);

  // K5: xres = x + o @ w_proj
  k_gemm_f32<<<dim3(768 / 64, (unsigned)(M / 64)), 128, 0, stream>>>(
      obf, wprojT, x, xres, (int)M, 768, 768);

  // K6: h2 = LN2(xres) -> bf16 (reuse hbf)
  k_layernorm<<<(unsigned)M, 256, 0, stream>>>(xres, n2_g, n2_b, hbf, 768);

  // K7: G = (h2@w_gate) * silu(h2@w_value) -> bf16
  k_swiglu<<<dim3(3072 / 64, (unsigned)(M / 64)), 128, 0, stream>>>(
      hbf, wgateT, wvalT, gbuf, (int)M, 3072, 768);

  // K8: out = xres + G @ w_out
  k_gemm_f32<<<dim3(768 / 64, (unsigned)(M / 64)), 128, 0, stream>>>(
      gbuf, woutT, xres, out, (int)M, 768, 3072);
}
